// FLABased_5351529251276
// MI455X (gfx1250) — compile-verified, matrix operand order checked by reading, not run on hardware
//
#include <hip/hip_runtime.h>

// ---------------------------------------------------------------------------
// FLA "Based" linear attention block for MI455X (gfx1250, wave32, WMMA).
// All matmuls run on v_wmma_f32_16x16x32_bf16. Projection GEMMs use
// double-buffered async global->LDS staging (ASYNCcnt) per CDNA5.
// GEMM wave tile 32x64 (2x4 frags): 8 WMMAs per 6 LDS fragment loads.
// ---------------------------------------------------------------------------

typedef __bf16 bf16_t;
typedef __attribute__((ext_vector_type(16))) __bf16 bf16x16;
typedef __attribute__((ext_vector_type(8)))  __bf16 bf16x8;
typedef __attribute__((ext_vector_type(4)))  __bf16 bf16x4;
typedef __attribute__((ext_vector_type(8)))  float  f32x8;

#define DEV_INLINE __device__ __forceinline__

static constexpr int BB = 4;     // batch
static constexpr int TT = 1024;  // sequence
static constexpr int DD = 1024;  // model dim
static constexpr int HH = 16;    // heads
static constexpr int HD = 64;    // head dim

// LDS tile geometry for the GEMM: BM=128, BN=128, BK=64, rows padded to 72
// elems (144B) so ds_load_b128 fragment reads hit distinct bank groups.
static constexpr int LDA = 72;   // padded row length (elements)

// --- gfx1250 async global->LDS copy (ASYNCcnt) ------------------------------
DEV_INLINE void async_ld_b128(unsigned lds_off, const void* gptr) {
  asm volatile("global_load_async_to_lds_b128 %0, %1, off"
               :: "v"(lds_off), "v"(gptr) : "memory");
}
DEV_INLINE void wait_async_le8() { asm volatile("s_wait_asynccnt 0x8" ::: "memory"); }
DEV_INLINE void wait_async_0()   { asm volatile("s_wait_asynccnt 0x0" ::: "memory"); }

// --- WMMA 16x16x32 bf16 fragment loader -------------------------------------
// A-fragment (16xK tile, ISA 7.12.2): lane L: row M = L&15, hi = L>>4
//   e[0..7]  = src[M][k0 + hi*8 + j],  e[8..15] = src[M][k0 + 16 + hi*8 + j]
// Same pattern with r0 = column-block loads the B-fragment of a matrix whose
// transpose is stored row-major.
DEV_INLINE bf16x16 load_frag_row(const bf16_t* __restrict__ src, int ld,
                                 int r0, int k0, int lane) {
  const int m = lane & 15, hi = lane >> 4;
  const bf16_t* p = src + (size_t)(r0 + m) * ld + k0 + hi * 8;
  bf16x8 lo = *(const bf16x8*)(p);        // 16B -> b128 load
  bf16x8 hb = *(const bf16x8*)(p + 16);   // 16B -> b128 load
  bf16x16 f;
#pragma unroll
  for (int j = 0; j < 8; ++j) { f[j] = lo[j]; f[8 + j] = hb[j]; }
  return f;
}

DEV_INLINE f32x8 wmma_bf16(const bf16x16& a, const bf16x16& b, const f32x8& c) {
  return __builtin_amdgcn_wmma_f32_16x16x32_bf16(false, a, false, b,
                                                 (short)0, c, false, false);
}

// --- f32 -> bf16 conversion (x4 vectorized) ----------------------------------
__global__ void cvt_f32_bf16(const float* __restrict__ in,
                             bf16_t* __restrict__ out, int n) {
  int i = (blockIdx.x * blockDim.x + threadIdx.x) * 4;
  if (i < n) {
    float4 v = *(const float4*)(in + i);
    bf16x4 o;
    o[0] = (bf16_t)v.x; o[1] = (bf16_t)v.y; o[2] = (bf16_t)v.z; o[3] = (bf16_t)v.w;
    *(bf16x4*)(out + i) = o;
  }
}

// --- LDS-tiled WMMA GEMM: Y(MxN) = A(MxK) @ W(NxK)^T ------------------------
// Block = 256 threads (8 waves, 4x2), tile 128x128, BK=64, double-buffered
// async global->LDS staging. Wave computes a 32x64 sub-tile (2x4 fragments).
// mode 0: bf16 scatter (B,H,T,hd)    (q/k projections)
// mode 2: bf16 scatter (B,H,hd,T)    (v projection, transposed for PV B-frags)
// mode 1: f32 row-major MxN          (final output projection)
__global__ __launch_bounds__(256) void gemm_bf16_wmma(
    const bf16_t* __restrict__ A, const bf16_t* __restrict__ W,
    bf16_t* __restrict__ out_bf, float* __restrict__ out_f32,
    int M, int N, int K, int mode) {
  __shared__ __align__(16) bf16_t sA[2][128 * LDA];  // 2 x 18KB
  __shared__ __align__(16) bf16_t sW[2][128 * LDA];  // 2 x 18KB

  const int tid  = threadIdx.x;
  const int lane = tid & 31;
  const int wv   = tid >> 5;
  const int hi   = lane >> 4;
  const int nlo  = lane & 15;
  const int wm   = wv >> 1;                 // 0..3  (M direction, 32 rows)
  const int wn   = wv & 1;                  // 0..1  (N direction, 64 cols)
  const int rowB = blockIdx.y * 128;        // block row base
  const int colB = blockIdx.x * 128;        // block col base

  const unsigned aBase = (unsigned)(size_t)&sA[0][0];
  const unsigned wBase = (unsigned)(size_t)&sW[0][0];
  const unsigned bufSz = (unsigned)(128 * LDA * sizeof(bf16_t));

  // Issue one BK=64 tile: A and W each 128x64 = 1024 x 16B chunks (4/thread).
  auto issue_tile = [&](int k0, int buf) {
#pragma unroll
    for (int i = 0; i < 4; ++i) {
      int ch = tid + 256 * i, row = ch >> 3, c16 = ch & 7;
      async_ld_b128(aBase + buf * bufSz + row * (LDA * 2) + c16 * 16,
                    A + (size_t)(rowB + row) * K + k0 + c16 * 8);
    }
#pragma unroll
    for (int i = 0; i < 4; ++i) {
      int ch = tid + 256 * i, row = ch >> 3, c16 = ch & 7;
      async_ld_b128(wBase + buf * bufSz + row * (LDA * 2) + c16 * 16,
                    W + (size_t)(colB + row) * K + k0 + c16 * 8);
    }
  };

  f32x8 acc[2][4] = {};
  const int nk = K / 64;
  issue_tile(0, 0);

  for (int it = 0; it < nk; ++it) {
    const int buf = it & 1;
    const bool pre = (it + 1 < nk);
    if (pre) issue_tile((it + 1) * 64, buf ^ 1);   // overlap DMA with compute
    if (pre) wait_async_le8(); else wait_async_0();// current tile has landed
    __syncthreads();

    const bf16_t* lA = &sA[buf][0];
    const bf16_t* lW = &sW[buf][0];
#pragma unroll
    for (int kk = 0; kk < 64; kk += 32) {
      bf16x16 a0 = load_frag_row(lA, LDA, wm * 32,      kk, lane);
      bf16x16 a1 = load_frag_row(lA, LDA, wm * 32 + 16, kk, lane);
#pragma unroll
      for (int ni = 0; ni < 4; ++ni) {
        bf16x16 b = load_frag_row(lW, LDA, wn * 64 + ni * 16, kk, lane);
        acc[0][ni] = wmma_bf16(a0, b, acc[0][ni]);
        acc[1][ni] = wmma_bf16(a1, b, acc[1][ni]);
      }
    }
    __syncthreads();   // all reads done before buffer is re-filled
  }

  // C/D layout: element j, lane L -> row = j + 8*(L>>4), col = (L&15)
  const int t0 = rowB + wm * 32;
  const int n0 = colB + wn * 64;
#pragma unroll
  for (int mi = 0; mi < 2; ++mi) {
#pragma unroll
    for (int ni = 0; ni < 4; ++ni) {
#pragma unroll
      for (int j = 0; j < 8; ++j) {
        const int r = t0 + mi * 16 + j + 8 * hi;
        const int c = n0 + ni * 16 + nlo;
        const float v = acc[mi][ni][j];
        if (mode == 1) {
          out_f32[(size_t)r * N + c] = v;
        } else {
          const int b_ = r >> 10, t_ = r & (TT - 1);
          const int h_ = c >> 6,  d_ = c & (HD - 1);
          if (mode == 0)   // (B,H,T,hd)
            out_bf[(((size_t)b_ * HH + h_) * TT + t_) * HD + d_] = (bf16_t)v;
          else             // mode 2: (B,H,hd,T)  == V^T per head
            out_bf[(((size_t)b_ * HH + h_) * HD + d_) * TT + t_] = (bf16_t)v;
        }
      }
    }
  }
}

// --- Based causal attention (quadratic form with Taylor score map) ----------
// One wave handles 16 query rows of one (b,h). grid = (B*H, T/128), block 256.
// v is pre-transposed per head so PV B-fragments are contiguous b128 loads.
__global__ __launch_bounds__(256) void based_attn_wmma(
    const bf16_t* __restrict__ q, const bf16_t* __restrict__ k,
    const bf16_t* __restrict__ vt, bf16_t* __restrict__ o) {
  __shared__ __align__(16) bf16_t lds_p[8][16 * 32];  // per-wave P restage tile

  const int lane = threadIdx.x & 31;
  const int wv   = threadIdx.x >> 5;
  const int nlo  = lane & 15;
  const int hi   = lane >> 4;
  const int head = blockIdx.x;               // b*H + h
  const int b = head >> 4, h = head & 15;
  const int t0 = (blockIdx.y * 8 + wv) * 16;

  const bf16_t* Qh = q  + (size_t)head * TT * HD;
  const bf16_t* Kh = k  + (size_t)head * TT * HD;
  const bf16_t* Vt = vt + (size_t)head * HD * TT;   // (hd, T) row-major
  bf16_t* lp = &lds_p[wv][0];

  const bf16x16 aQ0 = load_frag_row(Qh, HD, t0, 0,  lane);
  const bf16x16 aQ1 = load_frag_row(Qh, HD, t0, 32, lane);

  f32x8 acc[4] = {};
  f32x8 den = {};

  for (int s0 = 0; s0 < t0 + 16; s0 += 32) {
#pragma unroll
    for (int half = 0; half < 2; ++half) {
      const int st = s0 + half * 16;
      f32x8 sc = {};
      bf16x16 bK0 = load_frag_row(Kh, HD, st, 0,  lane);  // B[d][s] = K[s][d]
      sc = wmma_bf16(aQ0, bK0, sc);
      bf16x16 bK1 = load_frag_row(Kh, HD, st, 32, lane);
      sc = wmma_bf16(aQ1, bK1, sc);
      // score map + causal mask + denominator partials, stage P in LDS
#pragma unroll
      for (int j = 0; j < 8; ++j) {
        const int tg = t0 + j + 8 * hi;
        const int sg = st + nlo;
        float s = sc[j] * 0.125f;                 // 1/sqrt(64)
        float p = fmaf(0.5f * s, s, s) + 1.0f;    // 1 + s + s^2/2
        p = (sg <= tg) ? p : 0.0f;
        den[j] += p;
        lp[(j + 8 * hi) * 32 + half * 16 + nlo] = (bf16_t)p;
      }
    }
    // P (C-layout) -> A-fragment via LDS, then num += P @ V
    bf16x16 aP = load_frag_row(lp, 32, 0, 0, lane);
#pragma unroll
    for (int nd = 0; nd < 4; ++nd) {
      bf16x16 bV = load_frag_row(Vt, TT, nd * 16, s0, lane);  // rows of V^T
      acc[nd] = wmma_bf16(aP, bV, acc[nd]);
    }
  }

  // Reduce row sums across the 16 lanes of each half-wave (xor<=8 stays in half)
#pragma unroll
  for (int j = 0; j < 8; ++j) {
    float d = den[j];
    d += __shfl_xor(d, 1, 32);
    d += __shfl_xor(d, 2, 32);
    d += __shfl_xor(d, 4, 32);
    d += __shfl_xor(d, 8, 32);
    den[j] = 1.0f / d;
  }

  // o laid out (B, T, H*hd) bf16 for the final projection GEMM
#pragma unroll
  for (int nd = 0; nd < 4; ++nd) {
#pragma unroll
    for (int j = 0; j < 8; ++j) {
      const int tg = t0 + j + 8 * hi;
      const int c  = h * HD + nd * 16 + nlo;
      o[((size_t)b * TT + tg) * DD + c] = (bf16_t)(acc[nd][j] * den[j]);
    }
  }
}

// ---------------------------------------------------------------------------
extern "C" void kernel_launch(void* const* d_in, const int* in_sizes, int n_in,
                              void* d_out, int out_size, void* d_ws, size_t ws_size,
                              hipStream_t stream) {
  const float* x  = (const float*)d_in[0];
  const float* Wq = (const float*)d_in[1];
  const float* Wk = (const float*)d_in[2];
  const float* Wv = (const float*)d_in[3];
  const float* Wo = (const float*)d_in[4];
  float* out = (float*)d_out;

  const size_t nx = (size_t)BB * TT * DD;  // 4M elements
  const size_t nw = (size_t)DD * DD;       // 1M elements

  char* ws = (char*)d_ws;
  bf16_t* xbf = (bf16_t*)ws;              ws += nx * sizeof(bf16_t);
  bf16_t* wqb = (bf16_t*)ws;              ws += nw * sizeof(bf16_t);
  bf16_t* wkb = (bf16_t*)ws;              ws += nw * sizeof(bf16_t);
  bf16_t* wvb = (bf16_t*)ws;              ws += nw * sizeof(bf16_t);
  bf16_t* wob = (bf16_t*)ws;              ws += nw * sizeof(bf16_t);
  bf16_t* qh  = (bf16_t*)ws;              ws += nx * sizeof(bf16_t);
  bf16_t* kh  = (bf16_t*)ws;              ws += nx * sizeof(bf16_t);
  bf16_t* vth = (bf16_t*)ws;              ws += nx * sizeof(bf16_t);
  bf16_t* obf = (bf16_t*)ws;              ws += nx * sizeof(bf16_t);

  // 1) convert activations + weights to bf16
  cvt_f32_bf16<<<dim3((unsigned)(nx / 4 / 256)), 256, 0, stream>>>(x,  xbf, (int)nx);
  cvt_f32_bf16<<<dim3((unsigned)(nw / 4 / 256)), 256, 0, stream>>>(Wq, wqb, (int)nw);
  cvt_f32_bf16<<<dim3((unsigned)(nw / 4 / 256)), 256, 0, stream>>>(Wk, wkb, (int)nw);
  cvt_f32_bf16<<<dim3((unsigned)(nw / 4 / 256)), 256, 0, stream>>>(Wv, wvb, (int)nw);
  cvt_f32_bf16<<<dim3((unsigned)(nw / 4 / 256)), 256, 0, stream>>>(Wo, wob, (int)nw);

  // 2) q/k/v projections (q,k -> (B,H,T,hd); v -> (B,H,hd,T) transposed)
  const int M = BB * TT;                           // 4096
  dim3 ggrid(DD / 128, M / 128);                   // (8, 32)
  gemm_bf16_wmma<<<ggrid, 256, 0, stream>>>(xbf, wqb, qh,  nullptr, M, DD, DD, 0);
  gemm_bf16_wmma<<<ggrid, 256, 0, stream>>>(xbf, wkb, kh,  nullptr, M, DD, DD, 0);
  gemm_bf16_wmma<<<ggrid, 256, 0, stream>>>(xbf, wvb, vth, nullptr, M, DD, DD, 2);

  // 3) causal based attention
  based_attn_wmma<<<dim3(BB * HH, TT / 128), 256, 0, stream>>>(qh, kh, vth, obf);

  // 4) output projection -> f32 result
  gemm_bf16_wmma<<<ggrid, 256, 0, stream>>>(obf, wob, nullptr, out, M, DD, DD, 1);
}
